// Gemma4Router_18537078850098
// MI455X (gfx1250) — matrix operand order repaired and checked by reading, not compiled, not run on hardware
//
#include <hip/hip_runtime.h>

typedef __attribute__((ext_vector_type(16))) _Float16 v16h;
typedef __attribute__((ext_vector_type(8)))  float    v8f;
typedef __attribute__((ext_vector_type(4)))  float    v4f;

constexpr int T = 32768;
constexpr int H = 2048;
constexpr int E = 64;
constexpr int TOKS_PER_WAVE  = 16;
constexpr int WAVES_PER_BLK  = 8;
constexpr int TOKS_PER_BLK   = TOKS_PER_WAVE * WAVES_PER_BLK;   // 128

// D = A(16x32 f16) x B(32x16 f16) + C(16x16 f32), four 16-expert tiles.
__global__ __launch_bounds__(256) void gemma_router_kernel(
    const float*    __restrict__ x,      // [T, H] fp32
    const float*    __restrict__ scale,  // [H]    fp32
    const _Float16* __restrict__ W,      // [E, H] fp16 (row-major, expert-major)
    const float*    __restrict__ pes,    // [E]    fp32
    int*            __restrict__ idx_out,// [T, 4] int32
    float*          __restrict__ w_out)  // [T, 4] fp32
{
    __shared__ float lds_logits[TOKS_PER_BLK * E];   // 32 KB

    const int lane = threadIdx.x & 31;
    const int wave = threadIdx.x >> 5;               // 0..7
    const int m    = lane & 15;                      // token-in-tile / expert column
    const int hi   = lane >> 4;                      // 0|1 : K-half owned by this lane
    const int koff = hi * 16;
    const int tokBase = blockIdx.x * TOKS_PER_BLK + wave * TOKS_PER_WAVE;

    const float* __restrict__ xrow = x + (size_t)(tokBase + m) * H;

    // ---- Pass 1: sum of squares over this lane's A-fragment elements ----
    float ss = 0.0f;
    for (int kb = 0; kb < H; kb += 32) {
        v4f a0 = *(const v4f*)(xrow + kb + koff);
        v4f a1 = *(const v4f*)(xrow + kb + koff + 4);
        v4f a2 = *(const v4f*)(xrow + kb + koff + 8);
        v4f a3 = *(const v4f*)(xrow + kb + koff + 12);
        #pragma unroll
        for (int j = 0; j < 4; ++j)
            ss += a0[j]*a0[j] + a1[j]*a1[j] + a2[j]*a2[j] + a3[j]*a3[j];
    }
    ss += __shfl_xor(ss, 16, 32);                    // lane pair covers the full row
    // s_m = rsqrt(mean(x^2)+eps) * H^-0.5
    const float s_m = rsqrtf(ss * (1.0f / 2048.0f) + 1e-6f) * 0.02209708691207961f;

    // ---- Pass 2: WMMA GEMM, logits[16 tok x 64 experts] in 4 accumulators ----
    v8f c0 = {}, c1 = {}, c2 = {}, c3 = {};
    const _Float16* __restrict__ wbase = W + (size_t)m * H + hi * 16;
    for (int kb = 0; kb < H; kb += 32) {
        v4f xa = *(const v4f*)(xrow + kb + koff);
        v4f xb = *(const v4f*)(xrow + kb + koff + 4);
        v4f xc = *(const v4f*)(xrow + kb + koff + 8);
        v4f xd = *(const v4f*)(xrow + kb + koff + 12);
        v4f sa = *(const v4f*)(scale + kb + koff);
        v4f sb = *(const v4f*)(scale + kb + koff + 4);
        v4f sc = *(const v4f*)(scale + kb + koff + 8);
        v4f sd = *(const v4f*)(scale + kb + koff + 12);

        v16h A;
        #pragma unroll
        for (int j = 0; j < 4; ++j) {
            A[j]      = (_Float16)(xa[j] * sa[j] * s_m);
            A[j + 4]  = (_Float16)(xb[j] * sb[j] * s_m);
            A[j + 8]  = (_Float16)(xc[j] * sc[j] * s_m);
            A[j + 12] = (_Float16)(xd[j] * sd[j] * s_m);
        }

        const _Float16* wp = wbase + kb;
        v16h B0 = *(const v16h*)(wp);
        v16h B1 = *(const v16h*)(wp + (size_t)16 * H);
        v16h B2 = *(const v16h*)(wp + (size_t)32 * H);
        v16h B3 = *(const v16h*)(wp + (size_t)48 * H);

        c0 = __builtin_amdgcn_wmma_f32_16x16x32_f16(false, A, false, B0, (short)0, c0, false, false);
        c1 = __builtin_amdgcn_wmma_f32_16x16x32_f16(false, A, false, B1, (short)0, c1, false, false);
        c2 = __builtin_amdgcn_wmma_f32_16x16x32_f16(false, A, false, B2, (short)0, c2, false, false);
        c3 = __builtin_amdgcn_wmma_f32_16x16x32_f16(false, A, false, B3, (short)0, c3, false, false);
    }

    // ---- Scatter logits to LDS (C layout: VGPR r -> M = hi*8 + r, N = m) ----
    {
        float* lg = lds_logits + (size_t)(wave * TOKS_PER_WAVE + hi * 8) * E + m;
        #pragma unroll
        for (int r = 0; r < 8; ++r) {
            lg[r * E +  0] = c0[r];
            lg[r * E + 16] = c1[r];
            lg[r * E + 32] = c2[r];
            lg[r * E + 48] = c3[r];
        }
    }
    __syncthreads();

    // ---- Top-4 + renormalized softmax weights (softmax Z cancels) ----
    if (lane < 16) {
        const float* lg = lds_logits + (size_t)(wave * TOKS_PER_WAVE + lane) * E;
        float b0 = -3.4e38f, b1 = -3.4e38f, b2 = -3.4e38f, b3 = -3.4e38f;
        int   i0 = 0, i1 = 0, i2 = 0, i3 = 0;
        #pragma unroll 4
        for (int e = 0; e < E; ++e) {
            float v = lg[e];
            if (v > b3) {
                if (v > b0)      { b3=b2; i3=i2; b2=b1; i2=i1; b1=b0; i1=i0; b0=v; i0=e; }
                else if (v > b1) { b3=b2; i3=i2; b2=b1; i2=i1; b1=v;  i1=e; }
                else if (v > b2) { b3=b2; i3=i2; b2=v;  i2=e; }
                else             { b3=v;  i3=e; }
            }
        }
        const float e0 = 1.0f;
        const float e1 = __expf(b1 - b0);
        const float e2 = __expf(b2 - b0);
        const float e3 = __expf(b3 - b0);
        const float inv = 1.0f / (e0 + e1 + e2 + e3);

        const int gt = tokBase + lane;
        idx_out[gt * 4 + 0] = i0;  w_out[gt * 4 + 0] = e0 * inv * pes[i0];
        idx_out[gt * 4 + 1] = i1;  w_out[gt * 4 + 1] = e1 * inv * pes[i1];
        idx_out[gt * 4 + 2] = i2;  w_out[gt * 4 + 2] = e2 * inv * pes[i2];
        idx_out[gt * 4 + 3] = i3;  w_out[gt * 4 + 3] = e3 * inv * pes[i3];
    }
}

extern "C" void kernel_launch(void* const* d_in, const int* in_sizes, int n_in,
                              void* d_out, int out_size, void* d_ws, size_t ws_size,
                              hipStream_t stream) {
    (void)in_sizes; (void)n_in; (void)out_size; (void)d_ws; (void)ws_size;
    const float*    x     = (const float*)d_in[0];
    const float*    scale = (const float*)d_in[1];
    const _Float16* W     = (const _Float16*)d_in[2];
    const float*    pes   = (const float*)d_in[3];
    // d_in[4] = top_k scalar (known: 4)

    int*   idx_out = (int*)d_out;                       // [T,4] int32 first
    float* w_out   = (float*)d_out + (size_t)T * 4;     // [T,4] fp32 second

    dim3 grid(T / TOKS_PER_BLK);    // 256 workgroups
    dim3 block(256);                // 8 wave32
    gemma_router_kernel<<<grid, block, 0, stream>>>(x, scale, W, pes, idx_out, w_out);
}
